// NERLSTM_CRF_7670811591358
// MI455X (gfx1250) — matrix operand order repaired and checked by reading, not compiled, not run on hardware
//
#include <hip/hip_runtime.h>
#include <stdint.h>

#define T_LEN 256
#define B_SZ  64
#define E_DIM 256
#define HD    256
#define G4    1024     // 4*HD gate width
#define K_TAG 9

typedef __attribute__((ext_vector_type(16))) __bf16 v16bf;
typedef __attribute__((ext_vector_type(8)))  float  v8f;

// CDNA5 hardware V_TANH_F32 TRANS op.
__device__ __forceinline__ float fast_tanh(float x) {
#if __has_builtin(__builtin_amdgcn_tanhf)
  return __builtin_amdgcn_tanhf(x);
#elif __has_builtin(__builtin_amdgcn_tanh_f32)
  return __builtin_amdgcn_tanh_f32(x);
#else
  float r;
  asm volatile("v_tanh_f32 %0, %1\n\tv_nop" : "=v"(r) : "v"(x));
  return r;
#endif
}
__device__ __forceinline__ float sigm(float x) {
  return __builtin_fmaf(0.5f, fast_tanh(0.5f * x), 0.5f);
}

// ---------------------------------------------------------------------------
// Pack torch-layout weight w[N][K] (f32) into WMMA-B-fragment-major bf16:
// flat index = ((nt*(K/32)+kb)*32 + lane)*16 + e, with n = nt*16 + (lane&15),
// k = kb*32 + (e>>3)*16 + (lane>>4)*8 + (e&7). B operands become one 32B/lane load.
// ---------------------------------------------------------------------------
__global__ void pack_w_kernel(const float* __restrict__ w, __bf16* __restrict__ out,
                              int N, int K) {
  int idx = blockIdx.x * blockDim.x + threadIdx.x;
  if (idx >= N * K) return;
  int e    = idx & 15;
  int lane = (idx >> 4) & 31;
  int rest = idx >> 9;
  int KB   = K >> 5;
  int kb   = rest % KB;
  int nt   = rest / KB;
  int n = nt * 16 + (lane & 15);
  int k = kb * 32 + ((e >> 3) << 4) + (lane >> 4) * 8 + (e & 7);
  out[idx] = (__bf16)w[(size_t)n * K + k];
}

// Same packing for w_out [9][512], zero-padded to one 16-wide N tile.
__global__ void pack_wout_kernel(const float* __restrict__ w, __bf16* __restrict__ out) {
  int idx = blockIdx.x * blockDim.x + threadIdx.x;
  if (idx >= 16 * 2 * HD) return;
  int e    = idx & 15;
  int lane = (idx >> 4) & 31;
  int kb   = idx >> 9;              // 0..15
  int n = lane & 15;
  int k = kb * 32 + ((e >> 3) << 4) + (lane >> 4) * 8 + (e & 7);
  out[idx] = (n < K_TAG) ? (__bf16)w[(size_t)n * (2 * HD) + k] : (__bf16)0.0f;
}

__global__ void bias_kernel(const float* __restrict__ a, const float* __restrict__ b,
                            float* __restrict__ o, int n) {
  int i = blockIdx.x * blockDim.x + threadIdx.x;
  if (i < n) o[i] = a[i] + b[i];
}

// Embedding gather + f32->bf16: out[(t*B+b)*E + e] = embed[x[b][t]][e]
__global__ void embed_kernel(const int* __restrict__ x, const float* __restrict__ emb,
                             __bf16* __restrict__ out) {
  int idx = blockIdx.x * blockDim.x + threadIdx.x;
  if (idx >= T_LEN * B_SZ * E_DIM) return;
  int e  = idx & (E_DIM - 1);
  int tb = idx >> 8;
  int b  = tb & (B_SZ - 1);
  int t  = tb >> 6;
  int tok = x[b * T_LEN + t];
  out[idx] = (__bf16)emb[(size_t)tok * E_DIM + e];
}

// ---------------------------------------------------------------------------
// Dense bf16 GEMM, C = A[M=16384][K=256] @ Bpacked -> C[M][N=1024] f32.
// One wave computes a 16x64 strip (4 accumulator tiles); K fully unrolled.
// ---------------------------------------------------------------------------
__global__ __launch_bounds__(256) void gemm_bf16_kernel(
    const __bf16* __restrict__ A, const __bf16* __restrict__ Bpk,
    float* __restrict__ C) {
  constexpr int M = T_LEN * B_SZ, N = G4, K = E_DIM, KB = K / 32;
  int lane = threadIdx.x & 31, wv = threadIdx.x >> 5;
  int hf = lane >> 4, ln = lane & 15;
  int strip  = blockIdx.x * 8 + wv;
  constexpr int mtiles = M >> 4;
  int mt = strip % mtiles;
  int ns = strip / mtiles;
  const v16bf* Bp = (const v16bf*)Bpk;
  v8f acc[4] = {{}, {}, {}, {}};
  int arow = mt * 16 + ln;
#pragma unroll
  for (int kb = 0; kb < KB; ++kb) {
    v16bf af;  // ISA 16-bit A 16x32 layout: two contiguous 16B runs per lane
#pragma unroll
    for (int e = 0; e < 16; ++e) {
      int v = e >> 1, lo = e & 1;
      int k = ((v < 4) ? 0 : 16) + hf * 8 + ((v & 3) << 1) + lo;
      af[e] = A[(size_t)arow * K + kb * 32 + k];
    }
#pragma unroll
    for (int j = 0; j < 4; ++j) {
      int nt = ns * 4 + j;
      v16bf bf = Bp[(size_t)(nt * KB + kb) * 32 + lane];
      acc[j] = __builtin_amdgcn_wmma_f32_16x16x32_bf16(
          false, af, false, bf, (short)0, acc[j], false, false);
    }
  }
#pragma unroll
  for (int j = 0; j < 4; ++j)
#pragma unroll
    for (int r = 0; r < 8; ++r)
      C[(size_t)(mt * 16 + hf * 8 + r) * N + (ns * 4 + j) * 16 + ln] = acc[j][r];
}

// ---------------------------------------------------------------------------
// Recurrent bf16-WMMA LSTM. grid = (4 batch-groups of 16, 2 directions),
// 512 threads = 16 waves; wave w owns hidden columns [16w,16w+16) for all 4
// gates. c stays in accumulator VGPRs for all T. h double-buffers through
// LDS in bf16 (one barrier/step). Gx reads software-pipelined one step
// ahead; t+2 slab prefetched. h_out stored in bf16 (feeds a bf16 GEMM).
// ---------------------------------------------------------------------------
__global__ __launch_bounds__(512) void lstm_kernel(
    const float* __restrict__ Gx_f, const float* __restrict__ Gx_b,
    const __bf16* __restrict__ whh_f, const __bf16* __restrict__ whh_b,
    const float* __restrict__ bias_f, const float* __restrict__ bias_b,
    __bf16* __restrict__ h_out) {
  __shared__ __bf16 hs[2][16 * HD];

  const int dir   = blockIdx.y;
  const int mbase = blockIdx.x * 16;
  const float* Gx   = dir ? Gx_b : Gx_f;
  const v16bf* Bp   = (const v16bf*)(dir ? whh_b : whh_f);
  const float* bias = dir ? bias_b : bias_f;

  int lane = threadIdx.x & 31, wv = threadIdx.x >> 5;
  int hf = lane >> 4, ln = lane & 15;
  int js = wv * 16;

  for (int i = threadIdx.x; i < 16 * HD; i += 512) {
    hs[0][i] = (__bf16)0.0f;
    hs[1][i] = (__bf16)0.0f;
  }

  float bz[4];
#pragma unroll
  for (int g = 0; g < 4; ++g) bz[g] = bias[g * HD + js + ln];

  v8f cst = {};

  // preload Gx for the first step
  float gxv[4][8];
  {
    int t0 = dir ? (T_LEN - 1) : 0;
    const float* gp = Gx + ((size_t)t0 * B_SZ + mbase) * G4;
#pragma unroll
    for (int g = 0; g < 4; ++g)
#pragma unroll
      for (int r = 0; r < 8; ++r)
        gxv[g][r] = gp[(size_t)(hf * 8 + r) * G4 + g * HD + js + ln];
  }
  __syncthreads();

  for (int ti = 0; ti < T_LEN; ++ti) {
    const __bf16* hread = hs[ti & 1];
    __bf16*       hwrite = hs[(ti & 1) ^ 1];
    int t = dir ? (T_LEN - 1 - ti) : ti;

    // Issue next step's Gx loads now; latency hides under the 32 WMMAs.
    float gxn[4][8];
    if (ti + 1 < T_LEN) {
      int tn = dir ? (t - 1) : (t + 1);
      const float* gp = Gx + ((size_t)tn * B_SZ + mbase) * G4;
#pragma unroll
      for (int g = 0; g < 4; ++g)
#pragma unroll
        for (int r = 0; r < 8; ++r)
          gxn[g][r] = gp[(size_t)(hf * 8 + r) * G4 + g * HD + js + ln];
      if (ti + 2 < T_LEN) {  // warm L2 for t+2 (global_prefetch_b8)
        int tp = dir ? (t - 2) : (t + 2);
        const float* pp = Gx + ((size_t)tp * B_SZ + mbase) * G4;
#pragma unroll
        for (int g = 0; g < 4; ++g)
          __builtin_prefetch(&pp[(size_t)(hf * 8) * G4 + g * HD + js + ln], 0, 1);
      }
    } else {
#pragma unroll
      for (int g = 0; g < 4; ++g)
#pragma unroll
        for (int r = 0; r < 8; ++r) gxn[g][r] = 0.0f;
    }

    // Gh = h_{t-1} @ w_hh^T for this wave's 4 gate tiles
    v8f acc[4] = {{}, {}, {}, {}};
#pragma unroll
    for (int kb = 0; kb < HD / 32; ++kb) {
      v16bf af;
#pragma unroll
      for (int e = 0; e < 16; ++e) {
        int v = e >> 1, lo = e & 1;
        int k = ((v < 4) ? 0 : 16) + hf * 8 + ((v & 3) << 1) + lo;
        af[e] = hread[ln * HD + kb * 32 + k];
      }
#pragma unroll
      for (int g = 0; g < 4; ++g) {
        int nt = (g * HD + js) >> 4;
        v16bf bf = Bp[(size_t)(nt * (HD / 32) + kb) * 32 + lane];
        acc[g] = __builtin_amdgcn_wmma_f32_16x16x32_bf16(
            false, af, false, bf, (short)0, acc[g], false, false);
      }
    }

    // gate math: every nonlinearity is one hardware v_tanh_f32
    v8f hnew;
#pragma unroll
    for (int r = 0; r < 8; ++r) {
      float gi = acc[0][r] + gxv[0][r] + bz[0];
      float gf = acc[1][r] + gxv[1][r] + bz[1];
      float gg = acc[2][r] + gxv[2][r] + bz[2];
      float go = acc[3][r] + gxv[3][r] + bz[3];
      float cv = sigm(gf) * cst[r] + sigm(gi) * fast_tanh(gg);
      cst[r] = cv;
      hnew[r] = sigm(go) * fast_tanh(cv);
    }
#pragma unroll
    for (int r = 0; r < 8; ++r) {
      int m = hf * 8 + r;
      __bf16 hb = (__bf16)hnew[r];
      hwrite[m * HD + js + ln] = hb;
      h_out[((size_t)t * B_SZ + mbase + m) * (2 * HD) + dir * HD + js + ln] = hb;
    }
    __syncthreads();  // protects buffer reused next step + LDS visibility
#pragma unroll
    for (int g = 0; g < 4; ++g)
#pragma unroll
      for (int r = 0; r < 8; ++r) gxv[g][r] = gxn[g][r];
  }
}

// ---------------------------------------------------------------------------
// Emissions via WMMA: em[M=16384][9] = H[M][512] @ w_out^T (padded to 16
// columns) + b_out. One wave per 16-row M-tile, 16 WMMA k-steps; lanes with
// n>=9 are padding and do not store.
// ---------------------------------------------------------------------------
__global__ __launch_bounds__(256) void emis_wmma_kernel(
    const __bf16* __restrict__ H, const __bf16* __restrict__ Wpk,
    const float* __restrict__ b_out, float* __restrict__ em) {
  constexpr int K = 2 * HD, KB = K / 32;  // 512, 16
  int lane = threadIdx.x & 31, wv = threadIdx.x >> 5;
  int hf = lane >> 4, ln = lane & 15;
  int mt = blockIdx.x * 8 + wv;  // 1024 m-tiles total
  const v16bf* Bp = (const v16bf*)Wpk;
  v8f acc = {};
  int arow = mt * 16 + ln;
#pragma unroll
  for (int kb = 0; kb < KB; ++kb) {
    v16bf af;
#pragma unroll
    for (int e = 0; e < 16; ++e) {
      int v = e >> 1, lo = e & 1;
      int k = ((v < 4) ? 0 : 16) + hf * 8 + ((v & 3) << 1) + lo;
      af[e] = H[(size_t)arow * K + kb * 32 + k];
    }
    v16bf bf = Bp[(size_t)kb * 32 + lane];
    acc = __builtin_amdgcn_wmma_f32_16x16x32_bf16(
        false, af, false, bf, (short)0, acc, false, false);
  }
  if (ln < K_TAG) {
    float bo = b_out[ln];
#pragma unroll
    for (int r = 0; r < 8; ++r)
      em[(size_t)(mt * 16 + hf * 8 + r) * K_TAG + ln] = acc[r] + bo;
  }
}

// Single-block Viterbi: 1024 threads = 64 batches x 16 lanes (j<9 active).
__global__ __launch_bounds__(1024) void viterbi_kernel(
    const float* __restrict__ em, const float* __restrict__ start,
    const float* __restrict__ endv, const float* __restrict__ trans,
    int* __restrict__ hist, int* __restrict__ path, float* __restrict__ score) {
  __shared__ float sc[B_SZ * 16];
  __shared__ float nsc[B_SZ * 16];
  __shared__ float str[K_TAG * K_TAG];
  int tid = threadIdx.x;
  if (tid < K_TAG * K_TAG) str[tid] = trans[tid];
  int b = tid >> 4, j = tid & 15;
  bool act = (j < K_TAG);
  if (act) sc[b * 16 + j] = start[j] + em[(size_t)b * K_TAG + j];
  __syncthreads();
  for (int t = 1; t < T_LEN; ++t) {
    if (act) {
      float best = -3.4e38f; int arg = 0;
      for (int i = 0; i < K_TAG; ++i) {
        float v = sc[b * 16 + i] + str[i * K_TAG + j];
        if (v > best) { best = v; arg = i; }
      }
      nsc[b * 16 + j] = best + em[((size_t)t * B_SZ + b) * K_TAG + j];
      hist[((size_t)t * B_SZ + b) * K_TAG + j] = arg;
    }
    __syncthreads();
    if (act) sc[b * 16 + j] = nsc[b * 16 + j];
    __syncthreads();
  }
  if (act) sc[b * 16 + j] += endv[j];
  __threadfence_block();
  __syncthreads();
  if (tid < B_SZ) {
    int bb = tid;
    float best = -3.4e38f; int last = 0;
    for (int jj = 0; jj < K_TAG; ++jj) {
      float v = sc[bb * 16 + jj];
      if (v > best) { best = v; last = jj; }
    }
    score[bb] = best;
    path[bb * T_LEN + (T_LEN - 1)] = last;
    for (int t = T_LEN - 2; t >= 0; --t) {
      last = hist[((size_t)(t + 1) * B_SZ + bb) * K_TAG + last];
      path[bb * T_LEN + t] = last;
    }
  }
}

extern "C" void kernel_launch(void* const* d_in, const int* in_sizes, int n_in,
                              void* d_out, int out_size, void* d_ws, size_t ws_size,
                              hipStream_t stream) {
  (void)in_sizes; (void)n_in; (void)out_size; (void)ws_size;
  const int*   x       = (const int*)  d_in[0];
  const float* embed   = (const float*)d_in[1];
  const float* w_ih_f  = (const float*)d_in[2];
  const float* w_hh_f  = (const float*)d_in[3];
  const float* b_ih_f  = (const float*)d_in[4];
  const float* b_hh_f  = (const float*)d_in[5];
  const float* w_ih_b  = (const float*)d_in[6];
  const float* w_hh_b  = (const float*)d_in[7];
  const float* b_ih_b  = (const float*)d_in[8];
  const float* b_hh_b  = (const float*)d_in[9];
  const float* w_out   = (const float*)d_in[10];
  const float* b_out   = (const float*)d_in[11];
  const float* c_start = (const float*)d_in[12];
  const float* c_end   = (const float*)d_in[13];
  const float* c_trans = (const float*)d_in[14];

  char* cur = (char*)d_ws;
  auto take = [&](size_t bytes) -> void* {
    void* p = (void*)cur;
    cur += (bytes + 255) & ~(size_t)255;
    return p;
  };
  __bf16* emb_bf  = (__bf16*)take((size_t)T_LEN * B_SZ * E_DIM * 2);
  __bf16* wih_pf  = (__bf16*)take((size_t)G4 * E_DIM * 2);
  __bf16* whh_pf  = (__bf16*)take((size_t)G4 * HD * 2);
  __bf16* wih_pb  = (__bf16*)take((size_t)G4 * E_DIM * 2);
  __bf16* whh_pb  = (__bf16*)take((size_t)G4 * HD * 2);
  __bf16* wout_pk = (__bf16*)take((size_t)16 * 2 * HD * 2);
  float*  bias_f  = (float*)take((size_t)G4 * 4);
  float*  bias_b  = (float*)take((size_t)G4 * 4);
  float*  Gx_f    = (float*)take((size_t)T_LEN * B_SZ * G4 * 4);
  float*  Gx_b    = (float*)take((size_t)T_LEN * B_SZ * G4 * 4);
  __bf16* h_cat   = (__bf16*)take((size_t)T_LEN * B_SZ * 2 * HD * 2);
  float*  em      = (float*)take((size_t)T_LEN * B_SZ * K_TAG * 4);
  int*    hist    = (int*)  take((size_t)T_LEN * B_SZ * K_TAG * 4);

  int*   path_out  = (int*)d_out;                          // [B][T] int32
  float* score_out = (float*)d_out + (size_t)B_SZ * T_LEN; // [B] f32

  pack_w_kernel<<<(G4 * E_DIM + 255) / 256, 256, 0, stream>>>(w_ih_f, wih_pf, G4, E_DIM);
  pack_w_kernel<<<(G4 * HD + 255) / 256, 256, 0, stream>>>(w_hh_f, whh_pf, G4, HD);
  pack_w_kernel<<<(G4 * E_DIM + 255) / 256, 256, 0, stream>>>(w_ih_b, wih_pb, G4, E_DIM);
  pack_w_kernel<<<(G4 * HD + 255) / 256, 256, 0, stream>>>(w_hh_b, whh_pb, G4, HD);
  pack_wout_kernel<<<(16 * 2 * HD + 255) / 256, 256, 0, stream>>>(w_out, wout_pk);
  bias_kernel<<<(G4 + 255) / 256, 256, 0, stream>>>(b_ih_f, b_hh_f, bias_f, G4);
  bias_kernel<<<(G4 + 255) / 256, 256, 0, stream>>>(b_ih_b, b_hh_b, bias_b, G4);
  embed_kernel<<<(T_LEN * B_SZ * E_DIM + 255) / 256, 256, 0, stream>>>(x, embed, emb_bf);

  int strips = (T_LEN * B_SZ / 16) * (G4 / 64);  // 16384, exact multiple of 8
  gemm_bf16_kernel<<<strips / 8, 256, 0, stream>>>(emb_bf, wih_pf, Gx_f);
  gemm_bf16_kernel<<<strips / 8, 256, 0, stream>>>(emb_bf, wih_pb, Gx_b);

  dim3 lgrid(B_SZ / 16, 2);
  lstm_kernel<<<lgrid, 512, 0, stream>>>(Gx_f, Gx_b, whh_pf, whh_pb, bias_f, bias_b, h_cat);

  int mtiles = T_LEN * B_SZ / 16;  // 1024, exact multiple of 8
  emis_wmma_kernel<<<mtiles / 8, 256, 0, stream>>>(h_cat, wout_pk, b_out, em);
  viterbi_kernel<<<1, 1024, 0, stream>>>(em, c_start, c_end, c_trans, hist, path_out, score_out);
}
